// AdaptiveDecoder_83562883711062
// MI455X (gfx1250) — matrix operand order repaired and checked by reading, not compiled, not run on hardware
//
#include <hip/hip_runtime.h>
#include <hip/hip_bf16.h>
#include <math.h>

#define B_ 128
#define L_ 196
#define ENC_ 512
#define E_ 512
#define D_ 512
#define A_ 512
#define V_ 10000
#define VP_ 10048   // V padded to a multiple of 64 for the branch-free WMMA GEMM
#define MAXLEN_ 26
#define T_ 25

typedef __bf16 bf16_t;
typedef __attribute__((ext_vector_type(16))) __bf16 v16bf;
typedef __attribute__((ext_vector_type(8)))  float  v8f;
typedef __attribute__((ext_vector_type(4)))  unsigned int v4u;

union FragU { v4u q[2]; v16bf v; };

__device__ __forceinline__ float sigf(float x) { return 1.0f / (1.0f + expf(-x)); }

// A-fragment (16x32 bf16): lane m = lane&15; kb = (lane>>4)*8; VGPR0-3 = K kb..kb+7,
// VGPR4-7 = K kb+16..kb+23  -> two b128 loads.
__device__ __forceinline__ void load_a(FragU& f, const bf16_t* __restrict__ pa, int k0) {
  f.q[0] = *(const v4u*)(pa + k0);
  f.q[1] = *(const v4u*)(pa + k0 + 16);
}
// B-fragment (32x16 bf16): lane n = lane&15; kb = (lane>>4)*8; VGPR0-3 = K kb..kb+7,
// VGPR4-7 = K kb+16..kb+23 -> two b128 loads from the N-major (pre-transposed) weight.
__device__ __forceinline__ void load_b(FragU& f, const bf16_t* __restrict__ pb, int k0) {
  f.q[0] = *(const v4u*)(pb + k0);
  f.q[1] = *(const v4u*)(pb + k0 + 16);
}

// ---------------- WMMA GEMM: C[M,N] = A[M,K](bf16) * Bt[N,K]^T(bf16) (+bias) (+=C) ----
// Host-enforced: M % 16 == 0, N % 64 == 0, K % 64 == 0, K >= 128,
// grid covers exactly (M/16)*(N/64) wave-jobs. No branches in the vector part,
// so EXEC is all-ones at every WMMA (ISA requirement) and no acc PHIs/spills.
// K-loop is software-pipelined with two fragment register sets so next-chunk
// VMEM loads are in flight while the current chunk's WMMAs execute.
__global__ void ad_gemm_wmma(const bf16_t* __restrict__ A, const bf16_t* __restrict__ Bt,
                             const float* __restrict__ bias, float* __restrict__ C,
                             int M, int N, int K, int ldc, int accum) {
  const int lane = threadIdx.x & 31;
  const int job  = (blockIdx.x * blockDim.x + threadIdx.x) >> 5;
  const int nGroups = N >> 6;
  const int mTile   = job / nGroups;
  const int nGroup  = job - mTile * nGroups;
  const int m0      = mTile << 4;
  const int n0base  = nGroup << 6;

  const bf16_t* pa = A + (size_t)(m0 + (lane & 15)) * K + ((lane >> 4) << 3);
  const bf16_t* pb[4];
#pragma unroll
  for (int i = 0; i < 4; ++i)
    pb[i] = Bt + (size_t)(n0base + (i << 4) + (lane & 15)) * K + ((lane >> 4) << 3);

  v8f acc[4];
#pragma unroll
  for (int i = 0; i < 4; ++i) acc[i] = (v8f){0.f,0.f,0.f,0.f,0.f,0.f,0.f,0.f};

  FragU a0, a1, b0[4], b1[4];
  load_a(a0, pa, 0);
#pragma unroll
  for (int i = 0; i < 4; ++i) load_b(b0[i], pb[i], 0);

  int k0 = 0;
  for (; k0 + 96 <= K; k0 += 64) {
    load_a(a1, pa, k0 + 32);
#pragma unroll
    for (int i = 0; i < 4; ++i) load_b(b1[i], pb[i], k0 + 32);
#pragma unroll
    for (int i = 0; i < 4; ++i)
      acc[i] = __builtin_amdgcn_wmma_f32_16x16x32_bf16(
          false, a0.v, false, b0[i].v, (short)0, acc[i], false, false);
    load_a(a0, pa, k0 + 64);
#pragma unroll
    for (int i = 0; i < 4; ++i) load_b(b0[i], pb[i], k0 + 64);
#pragma unroll
    for (int i = 0; i < 4; ++i)
      acc[i] = __builtin_amdgcn_wmma_f32_16x16x32_bf16(
          false, a1.v, false, b1[i].v, (short)0, acc[i], false, false);
  }
  // tail: chunks k0 and k0+32 (K % 64 == 0 guarantees exactly two remain)
  load_a(a1, pa, k0 + 32);
#pragma unroll
  for (int i = 0; i < 4; ++i) load_b(b1[i], pb[i], k0 + 32);
#pragma unroll
  for (int i = 0; i < 4; ++i)
    acc[i] = __builtin_amdgcn_wmma_f32_16x16x32_bf16(
        false, a0.v, false, b0[i].v, (short)0, acc[i], false, false);
#pragma unroll
  for (int i = 0; i < 4; ++i)
    acc[i] = __builtin_amdgcn_wmma_f32_16x16x32_bf16(
        false, a1.v, false, b1[i].v, (short)0, acc[i], false, false);

  // C/D layout: lane n = lane&15; VGPR v -> m = v + ((lane>>4)<<3)
  const int nC    = lane & 15;
  const int mBase = (lane >> 4) << 3;
#pragma unroll
  for (int i = 0; i < 4; ++i) {
    const int n0 = n0base + (i << 4);
    float bv = bias ? bias[n0 + nC] : 0.0f;
#pragma unroll
    for (int v = 0; v < 8; ++v) {
      size_t idx = (size_t)(m0 + mBase + v) * ldc + (size_t)(n0 + nC);
      float val = acc[i][v] + bv;
      if (accum) val += C[idx];
      C[idx] = val;
    }
  }
}

// ---------------- helper/elementwise kernels ---------------------------------

__global__ void ad_f32_to_bf16(const float* __restrict__ in, bf16_t* __restrict__ out, int n) {
  int i = blockIdx.x * blockDim.x + threadIdx.x;
  if (i < n) out[i] = (bf16_t)in[i];
}

// W is K x N row-major (f32); Wt becomes Npad x K row-major (bf16), zero pad rows.
__global__ void ad_transpose_bf16(const float* __restrict__ W, bf16_t* __restrict__ Wt,
                                  int K, int N, int Npad) {
  int i = blockIdx.x * blockDim.x + threadIdx.x;
  if (i >= K * Npad) return;
  int n = i / K, k = i % K;
  float v = (n < N) ? W[(size_t)k * N + n] : 0.0f;
  Wt[i] = (bf16_t)v;
}

__global__ void ad_mean(const float* __restrict__ enc, float* __restrict__ mean) {
  int i = blockIdx.x * blockDim.x + threadIdx.x;
  if (i >= B_ * ENC_) return;
  int b = i / ENC_, e = i % ENC_;
  const float* p = enc + (size_t)b * L_ * ENC_ + e;
  float s = 0.f;
  for (int l = 0; l < L_; ++l) s += p[(size_t)l * ENC_];
  mean[i] = s * (1.0f / (float)L_);
}

__global__ void ad_init_hc(const float* __restrict__ mean,
                           const float* __restrict__ Wh, const float* __restrict__ bh,
                           const float* __restrict__ Wc, const float* __restrict__ bc,
                           float* __restrict__ h, float* __restrict__ c,
                           bf16_t* __restrict__ h_bf) {
  int i = blockIdx.x * blockDim.x + threadIdx.x;
  if (i >= B_ * D_) return;
  int b = i / D_, d = i % D_;
  const float* m = mean + (size_t)b * ENC_;
  float sh = bh[d], sc = bc[d];
  for (int k = 0; k < ENC_; ++k) {
    float mv = m[k];
    sh += mv * Wh[(size_t)k * D_ + d];
    sc += mv * Wc[(size_t)k * D_ + d];
  }
  float hv = tanhf(sh), cv = tanhf(sc);
  h[i] = hv; c[i] = cv; h_bf[i] = (bf16_t)hv;
}

__global__ void ad_gather_x(const float* __restrict__ emb, const int* __restrict__ captions,
                            int t, float* __restrict__ x, bf16_t* __restrict__ x_bf) {
  int i = blockIdx.x * blockDim.x + threadIdx.x;
  if (i >= B_ * E_) return;
  int b = i / E_, e = i % E_;
  int w = captions[b * MAXLEN_ + t];
  float v = emb[(size_t)w * E_ + e];
  x[i] = v; x_bf[i] = (bf16_t)v;
}

__global__ void ad_lstm(const float* __restrict__ gates, const float* __restrict__ s_pre,
                        const float* __restrict__ c_prev,
                        float* __restrict__ c_new, float* __restrict__ h_new,
                        bf16_t* __restrict__ h_bf, float* __restrict__ s,
                        bf16_t* __restrict__ s_bf) {
  int i = blockIdx.x * blockDim.x + threadIdx.x;
  if (i >= B_ * D_) return;
  int b = i / D_, d = i % D_;
  const float* g = gates + (size_t)b * 4 * D_;
  float ig = sigf(g[d]);
  float fg = sigf(g[D_ + d]);
  float gg = tanhf(g[2 * D_ + d]);
  float og = sigf(g[3 * D_ + d]);
  float cn = fg * c_prev[i] + ig * gg;
  float tc = tanhf(cn);
  float hn = og * tc;
  float sv = sigf(s_pre[i]) * tc;
  c_new[i] = cn; h_new[i] = hn; h_bf[i] = (bf16_t)hn;
  s[i] = sv; s_bf[i] = (bf16_t)sv;
}

// one wave per (b,l): e_v[b,l] = sum_a tanh(att_v + ag) * w_a
__global__ void ad_attn_ev(const float* __restrict__ att_v, const float* __restrict__ ag,
                           const float* __restrict__ w_a, float* __restrict__ e_v) {
  int wv = blockIdx.x * (blockDim.x >> 5) + (threadIdx.x >> 5);
  if (wv >= B_ * L_) return;
  int lane = threadIdx.x & 31;
  int b = wv / L_;
  const float* av  = att_v + (size_t)wv * A_;
  const float* agb = ag + (size_t)b * A_;
  float sum = 0.f;
  for (int a = lane; a < A_; a += 32) sum += tanhf(av[a] + agb[a]) * w_a[a];
  for (int off = 16; off; off >>= 1) sum += __shfl_down(sum, off, 32);
  if (lane == 0) e_v[wv] = sum;
}

// one wave per b: e_s[b] = sum_a tanh(es_pre + ag) * w_a
__global__ void ad_es(const float* __restrict__ es_pre, const float* __restrict__ ag,
                      const float* __restrict__ w_a, float* __restrict__ e_s) {
  int wv = blockIdx.x * (blockDim.x >> 5) + (threadIdx.x >> 5);
  if (wv >= B_) return;
  int lane = threadIdx.x & 31;
  const float* p = es_pre + (size_t)wv * A_;
  const float* q = ag + (size_t)wv * A_;
  float sum = 0.f;
  for (int a = lane; a < A_; a += 32) sum += tanhf(p[a] + q[a]) * w_a[a];
  for (int off = 16; off; off >>= 1) sum += __shfl_down(sum, off, 32);
  if (lane == 0) e_s[wv] = sum;
}

// one block per b: softmax over L+1, write masked alpha/beta, compute ctx
__global__ void ad_softmax_ctx(const float* __restrict__ e_v, const float* __restrict__ e_s,
                               const float* __restrict__ s, const float* __restrict__ enc,
                               const int* __restrict__ cap_len, int t,
                               float* __restrict__ alphas, float* __restrict__ betas,
                               float* __restrict__ ctx, bf16_t* __restrict__ ctx_bf) {
  __shared__ float sh[L_ + 1];
  __shared__ float red[2];
  int b = blockIdx.x;
  for (int i = threadIdx.x; i < L_; i += blockDim.x) sh[i] = e_v[(size_t)b * L_ + i];
  if (threadIdx.x == 0) sh[L_] = e_s[b];
  __syncthreads();
  if (threadIdx.x == 0) {
    float mx = sh[0];
    for (int i = 1; i <= L_; ++i) mx = fmaxf(mx, sh[i]);
    red[0] = mx;
  }
  __syncthreads();
  float mx = red[0];
  for (int i = threadIdx.x; i <= L_; i += blockDim.x) sh[i] = expf(sh[i] - mx);
  __syncthreads();
  if (threadIdx.x == 0) {
    float sm = 0.f;
    for (int i = 0; i <= L_; ++i) sm += sh[i];
    red[1] = 1.0f / sm;
  }
  __syncthreads();
  float inv = red[1];
  for (int i = threadIdx.x; i <= L_; i += blockDim.x) sh[i] *= inv;
  __syncthreads();

  int dl = cap_len[b] - 1; if (dl < 1) dl = 1;
  float active = (t < dl) ? 1.0f : 0.0f;
  for (int i = threadIdx.x; i < L_; i += blockDim.x)
    alphas[((size_t)b * T_ + t) * L_ + i] = sh[i] * active;
  if (threadIdx.x == 0) betas[(size_t)b * T_ + t] = sh[L_] * active;

  float beta = sh[L_];
  for (int e = threadIdx.x; e < ENC_; e += blockDim.x) {
    float acc = 0.f;
    const float* pe = enc + (size_t)b * L_ * ENC_ + e;
    for (int l = 0; l < L_; ++l) acc += sh[l] * pe[(size_t)l * ENC_];
    acc += beta * s[(size_t)b * D_ + e];
    ctx[(size_t)b * ENC_ + e] = acc;
    ctx_bf[(size_t)b * ENC_ + e] = (bf16_t)acc;
  }
}

__global__ void ad_add_convert(const float* __restrict__ u, const float* __restrict__ x,
                               bf16_t* __restrict__ u_bf, int n) {
  int i = blockIdx.x * blockDim.x + threadIdx.x;
  if (i < n) u_bf[i] = (bf16_t)(u[i] + x[i]);
}

// copy padded logits (B x VP_) into preds[:, t, :] (B x T x V) adding b_Lo
__global__ void ad_copy_logits(const float* __restrict__ lp, const float* __restrict__ b_Lo,
                               int t, float* __restrict__ preds) {
  int i = blockIdx.x * blockDim.x + threadIdx.x;
  if (i >= B_ * V_) return;
  int b = i / V_, v = i % V_;
  preds[((size_t)b * T_ + t) * V_ + v] = lp[(size_t)b * VP_ + v] + b_Lo[v];
}

// ---------------- host orchestration ----------------------------------------

extern "C" void kernel_launch(void* const* d_in, const int* in_sizes, int n_in,
                              void* d_out, int out_size, void* d_ws, size_t ws_size,
                              hipStream_t stream) {
  (void)in_sizes; (void)n_in; (void)out_size; (void)ws_size;
  const float* enc      = (const float*)d_in[0];
  const int*   captions = (const int*)d_in[1];
  const int*   cap_len  = (const int*)d_in[2];
  const float* emb      = (const float*)d_in[3];
  const float* W_ih     = (const float*)d_in[4];
  const float* W_hh     = (const float*)d_in[5];
  const float* b_lstm   = (const float*)d_in[6];
  const float* Wx_s     = (const float*)d_in[7];
  const float* Wh_s     = (const float*)d_in[8];
  const float* b_s      = (const float*)d_in[9];
  const float* W_v      = (const float*)d_in[10];
  const float* b_v      = (const float*)d_in[11];
  const float* W_g      = (const float*)d_in[12];
  const float* b_g      = (const float*)d_in[13];
  const float* W_s_att  = (const float*)d_in[14];
  const float* b_sa     = (const float*)d_in[15];
  const float* w_a      = (const float*)d_in[16];
  const float* W_init_h = (const float*)d_in[17];
  const float* b_init_h = (const float*)d_in[18];
  const float* W_init_c = (const float*)d_in[19];
  const float* b_init_c = (const float*)d_in[20];
  const float* W_Lh     = (const float*)d_in[21];
  const float* b_Lh     = (const float*)d_in[22];
  const float* W_Lz     = (const float*)d_in[23];
  const float* b_Lz     = (const float*)d_in[24];
  const float* W_Lo     = (const float*)d_in[25];
  const float* b_Lo     = (const float*)d_in[26];

  float* preds  = (float*)d_out;
  float* alphas = preds + (size_t)B_ * T_ * V_;
  float* betas  = alphas + (size_t)B_ * T_ * L_;

  size_t off = 0;
  auto alloc = [&](size_t bytes) -> void* {
    off = (off + 255) & ~(size_t)255;
    void* p = (char*)d_ws + off;
    off += bytes;
    return p;
  };
  bf16_t* enc_bf = (bf16_t*)alloc((size_t)B_ * L_ * ENC_ * 2);
  float*  att_v  = (float*) alloc((size_t)B_ * L_ * A_ * 4);
  bf16_t* WihT = (bf16_t*)alloc((size_t)4 * D_ * E_ * 2);
  bf16_t* WhhT = (bf16_t*)alloc((size_t)4 * D_ * D_ * 2);
  bf16_t* WxsT = (bf16_t*)alloc((size_t)D_ * E_ * 2);
  bf16_t* WhsT = (bf16_t*)alloc((size_t)D_ * D_ * 2);
  bf16_t* WvT  = (bf16_t*)alloc((size_t)A_ * ENC_ * 2);
  bf16_t* WgT  = (bf16_t*)alloc((size_t)A_ * D_ * 2);
  bf16_t* WsaT = (bf16_t*)alloc((size_t)A_ * ENC_ * 2);
  bf16_t* WLhT = (bf16_t*)alloc((size_t)E_ * D_ * 2);
  bf16_t* WLzT = (bf16_t*)alloc((size_t)E_ * ENC_ * 2);
  bf16_t* WLoT = (bf16_t*)alloc((size_t)VP_ * E_ * 2);
  float* meanb  = (float*)alloc((size_t)B_ * ENC_ * 4);
  float* hb0    = (float*)alloc((size_t)B_ * D_ * 4);
  float* hb1    = (float*)alloc((size_t)B_ * D_ * 4);
  float* cb0    = (float*)alloc((size_t)B_ * D_ * 4);
  float* cb1    = (float*)alloc((size_t)B_ * D_ * 4);
  bf16_t* hbf0  = (bf16_t*)alloc((size_t)B_ * D_ * 2);
  bf16_t* hbf1  = (bf16_t*)alloc((size_t)B_ * D_ * 2);
  float*  x     = (float*) alloc((size_t)B_ * E_ * 4);
  bf16_t* x_bf  = (bf16_t*)alloc((size_t)B_ * E_ * 2);
  float*  gatesb= (float*) alloc((size_t)B_ * 4 * D_ * 4);
  float*  s_pre = (float*) alloc((size_t)B_ * D_ * 4);
  float*  sb    = (float*) alloc((size_t)B_ * D_ * 4);
  bf16_t* s_bf  = (bf16_t*)alloc((size_t)B_ * D_ * 2);
  float*  ag    = (float*) alloc((size_t)B_ * A_ * 4);
  float*  e_v   = (float*) alloc((size_t)B_ * L_ * 4);
  float*  es_pre= (float*) alloc((size_t)B_ * A_ * 4);
  float*  e_s   = (float*) alloc((size_t)B_ * 4);
  float*  ctx   = (float*) alloc((size_t)B_ * ENC_ * 4);
  bf16_t* ctx_bf= (bf16_t*)alloc((size_t)B_ * ENC_ * 2);
  float*  ub    = (float*) alloc((size_t)B_ * E_ * 4);
  bf16_t* u_bf  = (bf16_t*)alloc((size_t)B_ * E_ * 2);
  float*  lgpad = (float*) alloc((size_t)B_ * VP_ * 4);

  float*  hbuf[2] = { hb0, hb1 };
  float*  cbuf[2] = { cb0, cb1 };
  bf16_t* hbf[2]  = { hbf0, hbf1 };

  const int TPB = 256;
  auto gsz = [](int n) { return (n + 255) / 256; };

  // N must be a multiple of 64, K a multiple of 64 (>=128); jobs = (M/16)*(N/64)
  // is a multiple of 4 for every call below, so the grid covers jobs exactly.
  auto launch_gemm = [&](const bf16_t* Aa, const bf16_t* Bt, const float* bias, float* C,
                         int M, int N, int K, int ldc, int accum) {
    int jobs = (M >> 4) * (N >> 6);
    int blocks = jobs / 4;  // 4 waves per block
    ad_gemm_wmma<<<blocks, 128, 0, stream>>>(Aa, Bt, bias, C, M, N, K, ldc, accum);
  };

  // ---- one-time prep (weights -> bf16 transposed; enc -> bf16) ----
  ad_f32_to_bf16<<<gsz(B_ * L_ * ENC_), TPB, 0, stream>>>(enc, enc_bf, B_ * L_ * ENC_);
  ad_transpose_bf16<<<gsz(E_ * 4 * D_), TPB, 0, stream>>>(W_ih, WihT, E_, 4 * D_, 4 * D_);
  ad_transpose_bf16<<<gsz(D_ * 4 * D_), TPB, 0, stream>>>(W_hh, WhhT, D_, 4 * D_, 4 * D_);
  ad_transpose_bf16<<<gsz(E_ * D_), TPB, 0, stream>>>(Wx_s, WxsT, E_, D_, D_);
  ad_transpose_bf16<<<gsz(D_ * D_), TPB, 0, stream>>>(Wh_s, WhsT, D_, D_, D_);
  ad_transpose_bf16<<<gsz(ENC_ * A_), TPB, 0, stream>>>(W_v, WvT, ENC_, A_, A_);
  ad_transpose_bf16<<<gsz(D_ * A_), TPB, 0, stream>>>(W_g, WgT, D_, A_, A_);
  ad_transpose_bf16<<<gsz(ENC_ * A_), TPB, 0, stream>>>(W_s_att, WsaT, ENC_, A_, A_);
  ad_transpose_bf16<<<gsz(D_ * E_), TPB, 0, stream>>>(W_Lh, WLhT, D_, E_, E_);
  ad_transpose_bf16<<<gsz(ENC_ * E_), TPB, 0, stream>>>(W_Lz, WLzT, ENC_, E_, E_);
  ad_transpose_bf16<<<gsz(E_ * VP_), TPB, 0, stream>>>(W_Lo, WLoT, E_, V_, VP_);

  ad_mean<<<gsz(B_ * ENC_), TPB, 0, stream>>>(enc, meanb);
  ad_init_hc<<<gsz(B_ * D_), TPB, 0, stream>>>(meanb, W_init_h, b_init_h, W_init_c, b_init_c,
                                               hbuf[0], cbuf[0], hbf[0]);

  // att_v = enc @ W_v + b_v  (25088 x 512 x 512)
  launch_gemm(enc_bf, WvT, b_v, att_v, B_ * L_, A_, ENC_, A_, 0);

  // ---- T sequential decode steps ----
  for (int t = 0; t < T_; ++t) {
    int cur = t & 1, nxt = cur ^ 1;
    ad_gather_x<<<gsz(B_ * E_), TPB, 0, stream>>>(emb, captions, t, x, x_bf);

    launch_gemm(x_bf,     WihT, b_lstm,  gatesb, B_, 4 * D_, E_, 4 * D_, 0);
    launch_gemm(hbf[cur], WhhT, nullptr, gatesb, B_, 4 * D_, D_, 4 * D_, 1);
    launch_gemm(x_bf,     WxsT, b_s,     s_pre,  B_, D_, E_, D_, 0);
    launch_gemm(hbf[cur], WhsT, nullptr, s_pre,  B_, D_, D_, D_, 1);

    ad_lstm<<<gsz(B_ * D_), TPB, 0, stream>>>(gatesb, s_pre, cbuf[cur],
                                              cbuf[nxt], hbuf[nxt], hbf[nxt], sb, s_bf);

    launch_gemm(hbf[nxt], WgT, b_g, ag, B_, A_, D_, A_, 0);
    ad_attn_ev<<<(B_ * L_ + 7) / 8, 256, 0, stream>>>(att_v, ag, w_a, e_v);
    launch_gemm(s_bf, WsaT, b_sa, es_pre, B_, A_, D_, A_, 0);
    ad_es<<<(B_ + 7) / 8, 256, 0, stream>>>(es_pre, ag, w_a, e_s);

    ad_softmax_ctx<<<B_, 256, 0, stream>>>(e_v, e_s, sb, enc, cap_len, t,
                                           alphas, betas, ctx, ctx_bf);

    launch_gemm(hbf[nxt], WLhT, b_Lh, ub, B_, E_, D_, E_, 0);
    launch_gemm(ctx_bf,   WLzT, b_Lz, ub, B_, E_, ENC_, E_, 1);
    ad_add_convert<<<gsz(B_ * E_), TPB, 0, stream>>>(ub, x, u_bf, B_ * E_);

    // logits (padded N) -> lgpad, then add bias + copy into preds[:, t, :]
    launch_gemm(u_bf, WLoT, nullptr, lgpad, B_, VP_, E_, VP_, 0);
    ad_copy_logits<<<gsz(B_ * V_), TPB, 0, stream>>>(lgpad, b_Lo, t, preds);
  }
}
